// RecurrentTransformer_45363444580916
// MI455X (gfx1250) — matrix operand order repaired and checked by reading, not compiled, not run on hardware
//
#include <hip/hip_runtime.h>
#include <hip/hip_bf16.h>

typedef __attribute__((ext_vector_type(16))) __bf16 v16bf;
typedef __attribute__((ext_vector_type(8)))  __bf16 v8bf;
typedef __attribute__((ext_vector_type(8)))  float  v8f;

#define LC   4
#define BBATCH 4
#define CINC 64
#define COUTC 64
#define EE   512
#define SSEQ 2048
#define MF   (BBATCH*SSEQ)

#define BM 128
#define BN 64
#define BK 32
#define LDT 40  // padded LDS row stride (bf16 elements)

// ---------------------------------------------------------------------------
// Elementwise helpers
// ---------------------------------------------------------------------------
__global__ void k_packw(const float* __restrict__ in, __bf16* __restrict__ out, int n) {
    int i = blockIdx.x * 256 + threadIdx.x;
    if (i < n) out[i] = (__bf16)in[i];
}

// cc_W [E][E][3] -> wpack [E][3*E] with wpack[o][d*E+i] = cc_W[o][i][d]
__global__ void k_packconv(const float* __restrict__ in, __bf16* __restrict__ out) {
    int i = blockIdx.x * 256 + threadIdx.x;
    if (i < EE * EE * 3) {
        int o = i / (3 * EE);
        int r = i % (3 * EE);
        int d = r / EE;
        int ii = r % EE;
        out[i] = (__bf16)in[(size_t)o * EE * 3 + (size_t)ii * 3 + d];
    }
}

// x [B][CIN][S] -> xt [B][S][CIN]
__global__ void k_xt(const float* __restrict__ x, float* __restrict__ xt) {
    int i = blockIdx.x * 256 + threadIdx.x;
    if (i < MF * CINC) {
        int b = i / (SSEQ * CINC);
        int r = i % (SSEQ * CINC);
        int s = r / CINC;
        int c = r % CINC;
        xt[i] = x[(size_t)b * CINC * SSEQ + (size_t)c * SSEQ + s];
    }
}

// LayerNorm over E=512, one wave per row, 4 rows per 128-thread block.
__global__ __launch_bounds__(128) void k_ln(const float* __restrict__ x,
                                            const float* __restrict__ g,
                                            const float* __restrict__ b,
                                            float* __restrict__ y) {
    int row  = blockIdx.x * 4 + (threadIdx.x >> 5);
    int lane = threadIdx.x & 31;
    const float* xr = x + (size_t)row * EE;
    float v[16];
    float s = 0.f, ss = 0.f;
#pragma unroll
    for (int i = 0; i < 16; ++i) {
        float f = xr[lane + i * 32];
        v[i] = f; s += f; ss += f * f;
    }
#pragma unroll
    for (int off = 16; off; off >>= 1) {
        s  += __shfl_xor(s,  off, 32);
        ss += __shfl_xor(ss, off, 32);
    }
    float mean = s * (1.f / EE);
    float var  = ss * (1.f / EE) - mean * mean;
    float inv  = rsqrtf(var + 1e-5f);
    float* yr = y + (size_t)row * EE;
#pragma unroll
    for (int i = 0; i < 16; ++i) {
        int e = lane + i * 32;
        yr[e] = (v[i] - mean) * inv * g[e] + b[e];
    }
}

// den[row] = sum over t of P[row][t] (P already tril-masked, bf16)
__global__ __launch_bounds__(256) void k_den(const __bf16* __restrict__ P,
                                             float* __restrict__ den) {
    __shared__ float red[256];
    int row = blockIdx.x;
    int t = threadIdx.x;
    const __bf16* pr = P + (size_t)row * SSEQ;
    float s = 0.f;
    for (int i = t; i < SSEQ; i += 256) s += (float)pr[i];
    red[t] = s;
    __syncthreads();
    for (int off = 128; off; off >>= 1) {
        if (t < off) red[t] += red[t + off];
        __syncthreads();
    }
    if (t == 0) den[row] = red[0];
}

// ---------------------------------------------------------------------------
// Universal bf16 WMMA GEMM:  C = epilogue(A @ Bw^T + bias [+ resid])
//   Block 128 threads (4 waves), tile 128x64, double-buffered LDS,
//   wave tile 64x32 -> 8 v_wmma_f32_16x16x32_bf16 per k-step per wave.
//   A: logical [M,K]; Bw: [N,K] row-major bf16 (ldb == K)
//   AMODE: 0 fp32 A (lda), 1 fp32 conv A (shifted rows, lda=E),
//          2 bf16 A, 3 fp32 A scaled by 1/(den[row]+1e-6)
//   ACT:   0 none, 1 phi=elu+1, 2 exact gelu
//   SMODE: 0 f32 [m*N+n], 1 bf16 [m*N+n] (opt tril mask),
//          2 bf16 transposed [b][n][s], 3 f32 transposed [b][n][s]
// ---------------------------------------------------------------------------
template<int AMODE, int ACT, int SMODE, bool RESID>
__global__ __launch_bounds__(128) void gemm_k(
    const void* __restrict__ A_, const __bf16* __restrict__ Bw_,
    const float* __restrict__ bias, const float* __restrict__ resid,
    const float* __restrict__ den, void* __restrict__ C_,
    int M, int N, int K, int lda,
    long long sA, long long sB, long long sC,
    int Sdim, int trilS, int trilA)
{
    __shared__ __bf16 As[2][BM * LDT];
    __shared__ __bf16 Bs[2][BN * LDT];

    const int tid  = threadIdx.x;
    const int lane = tid & 31;
    const int wv   = tid >> 5;
    const int wm   = (wv >> 1) * 64;   // wave M offset (2 waves in M)
    const int wn   = (wv & 1) * 32;    // wave N offset (2 waves in N)
    const int n0   = blockIdx.x * BN;
    const int m0   = blockIdx.y * BM;
    const int z    = blockIdx.z;

    const int lr = tid >> 1;        // 0..63
    const int lc = (tid & 1) * 16;  // k-chunk 0 or 16

    // fully-masked score tiles: zero-fill and exit
    if (SMODE == 1 && trilS && n0 >= m0 + BM) {
        __bf16* P = (__bf16*)C_ + (size_t)z * sC;
        uint4 zz; zz.x = zz.y = zz.z = zz.w = 0u;
#pragma unroll
        for (int h = 0; h < 2; ++h) {
            uint4* dst = (uint4*)(P + (size_t)(m0 + lr + h * 64) * N + (tid & 1) * 32);
            dst += (size_t)n0 / 8;  // n0 multiple of 64 -> stays 16B aligned
            dst[0] = zz; dst[1] = zz; dst[2] = zz; dst[3] = zz;
        }
        return;
    }

    const __bf16* Bw = Bw_ + (size_t)z * sB;

    // A fetch: 16 elements at (tile-row ar, k-col k0+lc), converted to bf16
    auto fetchA = [&](int ar, int k0, v8bf& lo, v8bf& hi) {
        if (AMODE == 2) {
            const __bf16* Ab = (const __bf16*)A_ + (size_t)z * sA;
            const v8bf* p = (const v8bf*)(Ab + (size_t)(m0 + ar) * lda + k0 + lc);
            lo = p[0]; hi = p[1];
        } else {
            const float* Af = (const float*)A_ + (size_t)z * sA;
            float f[16];
            bool valid = true;
            size_t base;
            if (AMODE == 1) {
                int kk = k0 + lc;
                int d  = kk >> 9;        // E = 512
                int ii = kk & 511;
                int ssrc = (m0 + ar) - 2 + d;
                valid = (ssrc >= 0);
                base = (size_t)(valid ? ssrc : 0) * (size_t)lda + ii;
            } else {
                base = (size_t)(m0 + ar) * (size_t)lda + (k0 + lc);
            }
            if (valid) {
                const float4* p = (const float4*)(Af + base);
                float4 x0 = p[0], x1 = p[1], x2 = p[2], x3 = p[3];
                f[0]=x0.x; f[1]=x0.y; f[2]=x0.z; f[3]=x0.w;
                f[4]=x1.x; f[5]=x1.y; f[6]=x1.z; f[7]=x1.w;
                f[8]=x2.x; f[9]=x2.y; f[10]=x2.z; f[11]=x2.w;
                f[12]=x3.x; f[13]=x3.y; f[14]=x3.z; f[15]=x3.w;
            } else {
#pragma unroll
                for (int i = 0; i < 16; ++i) f[i] = 0.f;
            }
            if (AMODE == 3) {
                float sc = 1.0f / (den[(size_t)m0 + ar] + 1e-6f);
#pragma unroll
                for (int i = 0; i < 16; ++i) f[i] *= sc;
            }
#pragma unroll
            for (int i = 0; i < 8; ++i) {
                lo[i] = (__bf16)f[i];
                hi[i] = (__bf16)f[i + 8];
            }
        }
    };
    auto fetchB = [&](int k0, v8bf& lo, v8bf& hi) {
        const v8bf* p = (const v8bf*)(Bw + (size_t)(n0 + lr) * K + k0 + lc);
        lo = p[0]; hi = p[1];
    };

    v8f acc[4][2];
#pragma unroll
    for (int i = 0; i < 4; ++i)
#pragma unroll
        for (int j = 0; j < 2; ++j)
#pragma unroll
            for (int e = 0; e < 8; ++e) acc[i][j][e] = 0.f;

    int ktiles = K / BK;
    if (trilA) {
        int kmax = m0 + BM;
        if (kmax > K) kmax = K;
        ktiles = kmax / BK;
    }

    // prologue: stage tile 0 into buffer 0
    {
        v8bf a0l, a0h, a1l, a1h, bl, bh;
        fetchA(lr,      0, a0l, a0h);
        fetchA(lr + 64, 0, a1l, a1h);
        fetchB(0, bl, bh);
        *(v8bf*)&As[0][lr * LDT + lc]            = a0l;
        *(v8bf*)&As[0][lr * LDT + lc + 8]        = a0h;
        *(v8bf*)&As[0][(lr + 64) * LDT + lc]     = a1l;
        *(v8bf*)&As[0][(lr + 64) * LDT + lc + 8] = a1h;
        *(v8bf*)&Bs[0][lr * LDT + lc]            = bl;
        *(v8bf*)&Bs[0][lr * LDT + lc + 8]        = bh;
    }

    const int am  = lane & 15;
    const int akb = (lane >> 4) * 8;   // A K-halves {0..7,16..23} / {8..15,24..31}
    const int bkb = (lane >> 4) * 8;   // B K-halves {0..7,16..23} / {8..15,24..31}

    for (int kt = 0; kt < ktiles; ++kt) {
        const int cur = kt & 1;
        __syncthreads();
        const bool hasNext = (kt + 1 < ktiles);
        v8bf na0l, na0h, na1l, na1h, nbl, nbh;
        if (hasNext) {                       // issue next tile's global loads
            const int kn = (kt + 1) * BK;
            fetchA(lr,      kn, na0l, na0h);
            fetchA(lr + 64, kn, na1l, na1h);
            fetchB(kn, nbl, nbh);
        }
        // compute on current buffer while loads are in flight
        v16bf aF[4], bF[2];
#pragma unroll
        for (int fm = 0; fm < 4; ++fm) {
            const int r = wm + fm * 16 + am;
            v8bf lo = *(const v8bf*)&As[cur][r * LDT + akb];
            v8bf hi = *(const v8bf*)&As[cur][r * LDT + akb + 16];
            aF[fm] = __builtin_shufflevector(lo, hi, 0,1,2,3,4,5,6,7,8,9,10,11,12,13,14,15);
        }
#pragma unroll
        for (int fn = 0; fn < 2; ++fn) {
            const int r = wn + fn * 16 + am;
            v8bf lo = *(const v8bf*)&Bs[cur][r * LDT + bkb];
            v8bf hi = *(const v8bf*)&Bs[cur][r * LDT + bkb + 16];
            bF[fn] = __builtin_shufflevector(lo, hi, 0,1,2,3,4,5,6,7,8,9,10,11,12,13,14,15);
        }
#pragma unroll
        for (int fm = 0; fm < 4; ++fm)
#pragma unroll
            for (int fn = 0; fn < 2; ++fn)
                acc[fm][fn] = __builtin_amdgcn_wmma_f32_16x16x32_bf16(
                    false, aF[fm], false, bF[fn], (short)0, acc[fm][fn], false, false);
        // stage next tile into alternate buffer
        if (hasNext) {
            const int nxt = cur ^ 1;
            *(v8bf*)&As[nxt][lr * LDT + lc]            = na0l;
            *(v8bf*)&As[nxt][lr * LDT + lc + 8]        = na0h;
            *(v8bf*)&As[nxt][(lr + 64) * LDT + lc]     = na1l;
            *(v8bf*)&As[nxt][(lr + 64) * LDT + lc + 8] = na1h;
            *(v8bf*)&Bs[nxt][lr * LDT + lc]            = nbl;
            *(v8bf*)&Bs[nxt][lr * LDT + lc + 8]        = nbh;
        }
    }

    // ---- epilogue
#pragma unroll
    for (int fm = 0; fm < 4; ++fm) {
#pragma unroll
        for (int fn = 0; fn < 2; ++fn) {
            const int mB = m0 + wm + fm * 16 + ((lane >> 4) << 3);
            const int nC = n0 + wn + fn * 16 + (lane & 15);
            const float bvv = bias ? bias[nC] : 0.f;
#pragma unroll
            for (int r = 0; r < 8; ++r) {
                const int m = mB + r;
                float val = acc[fm][fn][r] + bvv;
                if (ACT == 1)      val = (val > 0.f) ? (val + 1.f) : __expf(val);      // elu+1
                else if (ACT == 2) val = 0.5f * val * (1.f + erff(val * 0.70710678118f));
                if (RESID)         val += resid[(size_t)m * N + nC];
                if (SMODE == 0) {
                    ((float*)C_)[(size_t)z * sC + (size_t)m * N + nC] = val;
                } else if (SMODE == 1) {
                    if (trilS && nC > m) val = 0.f;
                    ((__bf16*)C_)[(size_t)z * sC + (size_t)m * N + nC] = (__bf16)val;
                } else if (SMODE == 2) {
                    int b = m / Sdim, s = m % Sdim;
                    ((__bf16*)C_)[((size_t)b * N + nC) * Sdim + s] = (__bf16)val;
                } else {
                    int b = m / Sdim, s = m % Sdim;
                    ((float*)C_)[((size_t)b * N + nC) * Sdim + s] = val;
                }
            }
        }
    }
}

// ---------------------------------------------------------------------------
// Host orchestration
// ---------------------------------------------------------------------------
extern "C" void kernel_launch(void* const* d_in, const int* in_sizes, int n_in,
                              void* d_out, int out_size, void* d_ws, size_t ws_size,
                              hipStream_t stream) {
    const float* x    = (const float*)d_in[0];
    const float* in_W = (const float*)d_in[1];
    const float* in_b = (const float*)d_in[2];
    const float* cc_W = (const float*)d_in[3];
    const float* cc_b = (const float*)d_in[4];
    const float* ln1g = (const float*)d_in[5];
    const float* ln1b = (const float*)d_in[6];
    const float* ln2g = (const float*)d_in[7];
    const float* ln2b = (const float*)d_in[8];
    const float* Wq = (const float*)d_in[9];   const float* bq = (const float*)d_in[10];
    const float* Wk = (const float*)d_in[11];  const float* bk = (const float*)d_in[12];
    const float* Wv = (const float*)d_in[13];  const float* bv = (const float*)d_in[14];
    const float* Wo = (const float*)d_in[15];  const float* bo = (const float*)d_in[16];
    const float* c1W = (const float*)d_in[17]; const float* c1b = (const float*)d_in[18];
    const float* c2W = (const float*)d_in[19]; const float* c2b = (const float*)d_in[20];
    const float* outW = (const float*)d_in[21]; const float* outb = (const float*)d_in[22];

    char* w = (char*)d_ws;
    auto alloc = [&](size_t bytes) {
        char* p = w;
        w += (bytes + 255) & ~(size_t)255;
        return (void*)p;
    };
    float*  lat   = (float*)alloc((size_t)MF * EE * 4);
    float*  mid   = (float*)alloc((size_t)MF * EE * 4);
    float*  hb    = (float*)alloc((size_t)MF * EE * 4);
    float*  qb    = (float*)alloc((size_t)MF * EE * 4);
    float*  t1    = (float*)alloc((size_t)MF * EE * 4);
    __bf16* kbf   = (__bf16*)alloc((size_t)MF * EE * 2);
    __bf16* vT    = (__bf16*)alloc((size_t)MF * EE * 2);
    __bf16* P     = (__bf16*)alloc((size_t)BBATCH * SSEQ * SSEQ * 2);
    float*  den   = (float*)alloc((size_t)MF * 4);
    __bf16* wpack = (__bf16*)alloc((size_t)EE * 3 * EE * 2);
    float*  xt    = (float*)alloc((size_t)MF * CINC * 4);

    const dim3 blk(128);
    const long long SE  = (long long)SSEQ * EE;
    const long long SS2 = (long long)SSEQ * SSEQ;

    // input layer: lat_pre = xt @ in_W.T + in_b      -> t1 [B,S,E]
    k_xt<<<(MF * CINC + 255) / 256, 256, 0, stream>>>(x, xt);
    k_packw<<<(EE * CINC + 255) / 256, 256, 0, stream>>>(in_W, wpack, EE * CINC);
    gemm_k<0,0,0,false><<<dim3(EE/BN, MF/BM, 1), blk, 0, stream>>>(
        xt, wpack, in_b, nullptr, nullptr, t1,
        MF, EE, CINC, CINC, 0, 0, 0, SSEQ, 0, 0);

    // causal conv (K = 3E, shifted-row A)            -> lat [B,S,E]
    k_packconv<<<(EE * EE * 3 + 255) / 256, 256, 0, stream>>>(cc_W, wpack);
    gemm_k<1,0,0,false><<<dim3(EE/BN, SSEQ/BM, BBATCH), blk, 0, stream>>>(
        t1, wpack, cc_b, nullptr, nullptr, lat,
        SSEQ, EE, 3 * EE, EE, SE, 0, SE, SSEQ, 0, 0);

    for (int l = 0; l < LC; ++l) {
        const size_t wo = (size_t)l * EE * EE;
        // LN1
        k_ln<<<MF / 4, 128, 0, stream>>>(lat, ln1g + l * EE, ln1b + l * EE, hb);
        // q = phi(h @ Wq.T + bq)  (fp32)
        k_packw<<<(EE*EE + 255)/256, 256, 0, stream>>>(Wq + wo, wpack, EE * EE);
        gemm_k<0,1,0,false><<<dim3(EE/BN, MF/BM, 1), blk, 0, stream>>>(
            hb, wpack, bq + l * EE, nullptr, nullptr, qb,
            MF, EE, EE, EE, 0, 0, 0, SSEQ, 0, 0);
        // k = phi(h @ Wk.T + bk)  (bf16 row-major: ready as scores B operand)
        k_packw<<<(EE*EE + 255)/256, 256, 0, stream>>>(Wk + wo, wpack, EE * EE);
        gemm_k<0,1,1,false><<<dim3(EE/BN, MF/BM, 1), blk, 0, stream>>>(
            hb, wpack, bk + l * EE, nullptr, nullptr, kbf,
            MF, EE, EE, EE, 0, 0, 0, SSEQ, 0, 0);
        // v = h @ Wv.T + bv  (bf16 transposed [B,E,S]: ready as num B operand)
        k_packw<<<(EE*EE + 255)/256, 256, 0, stream>>>(Wv + wo, wpack, EE * EE);
        gemm_k<0,0,2,false><<<dim3(EE/BN, MF/BM, 1), blk, 0, stream>>>(
            hb, wpack, bv + l * EE, nullptr, nullptr, vT,
            MF, EE, EE, EE, 0, 0, 0, SSEQ, 0, 0);
        // scores P = tril(q @ k.T)  bf16, batched; masked tiles zero-filled fast
        gemm_k<0,0,1,false><<<dim3(SSEQ/BN, SSEQ/BM, BBATCH), blk, 0, stream>>>(
            qb, kbf, nullptr, nullptr, nullptr, P,
            SSEQ, SSEQ, EE, EE, SE, SE, SS2, SSEQ, 1, 0);
        // den = rowsum(P)
        k_den<<<MF, 256, 0, stream>>>(P, den);
        // num = P @ v  (A bf16, K-loop clipped at the diagonal)
        gemm_k<2,0,0,false><<<dim3(EE/BN, SSEQ/BM, BBATCH), blk, 0, stream>>>(
            P, vT, nullptr, nullptr, nullptr, t1,
            SSEQ, EE, SSEQ, SSEQ, SS2, SE, SE, SSEQ, 0, 1);
        // mid = (num/(den+eps)) @ Wo.T + bo + lat
        k_packw<<<(EE*EE + 255)/256, 256, 0, stream>>>(Wo + wo, wpack, EE * EE);
        gemm_k<3,0,0,true><<<dim3(EE/BN, MF/BM, 1), blk, 0, stream>>>(
            t1, wpack, bo + l * EE, lat, den, mid,
            MF, EE, EE, EE, 0, 0, 0, SSEQ, 0, 0);
        // LN2
        k_ln<<<MF / 4, 128, 0, stream>>>(mid, ln2g + l * EE, ln2b + l * EE, hb);
        // mlp1 = gelu(h @ c1.T + c1b)
        k_packw<<<(EE*EE + 255)/256, 256, 0, stream>>>(c1W + wo, wpack, EE * EE);
        gemm_k<0,2,0,false><<<dim3(EE/BN, MF/BM, 1), blk, 0, stream>>>(
            hb, wpack, c1b + l * EE, nullptr, nullptr, qb,
            MF, EE, EE, EE, 0, 0, 0, SSEQ, 0, 0);
        // lat = mlp1 @ c2.T + c2b + mid
        k_packw<<<(EE*EE + 255)/256, 256, 0, stream>>>(c2W + wo, wpack, EE * EE);
        gemm_k<0,0,0,true><<<dim3(EE/BN, MF/BM, 1), blk, 0, stream>>>(
            qb, wpack, c2b + l * EE, mid, nullptr, lat,
            MF, EE, EE, EE, 0, 0, 0, SSEQ, 0, 0);
    }

    // output: out[b,o,s] = lat @ out_W.T + out_b  (transposed f32 store)
    k_packw<<<(COUTC * EE + 255) / 256, 256, 0, stream>>>(outW, wpack, COUTC * EE);
    gemm_k<0,0,3,false><<<dim3(COUTC/BN, MF/BM, 1), blk, 0, stream>>>(
        lat, wpack, outb, nullptr, nullptr, (float*)d_out,
        MF, COUTC, EE, EE, 0, 0, 0, SSEQ, 0, 0);
}
